// MHA_953482739759
// MI455X (gfx1250) — compile-verified, matrix operand order checked by reading, not run on hardware
//
#include <hip/hip_runtime.h>
#include <hip/hip_bf16.h>
#include <stdint.h>

#define HIDDEN 1024
#define HEADS  16
#define HDIM   64
#define BATCH  2
#define SEQ    2048
#define ROWS   (BATCH * SEQ)   // 4096

typedef __attribute__((ext_vector_type(16))) __bf16 v16bf;
typedef __attribute__((ext_vector_type(8)))  float  v8f;

struct Frag32 { uint4 lo, hi; };   // 32 bytes == 16 bf16

// WMMA bf16: D = A(16x32) * B(32x16) + C, fp32 accum
__device__ __forceinline__ v8f wmma_bf16(v16bf a, v16bf b, v8f c) {
    return __builtin_amdgcn_wmma_f32_16x16x32_bf16(
        /*neg_a=*/false, a, /*neg_b=*/false, b,
        /*c_mod=*/(short)0, c, /*reuse_a=*/false, /*reuse_b=*/false);
}

// float -> bf16 (round-to-nearest-even), bit pattern as ushort
__device__ __forceinline__ unsigned short f2bf(float x) {
    union { float f; unsigned u; } v; v.f = x;
    unsigned r = v.u + 0x7FFFu + ((v.u >> 16) & 1u);
    return (unsigned short)(r >> 16);
}

// Operand fragment (16x32 bf16). ISA layout: lane row m = lane&15,
// kb = (lane<16)?0:8; elems 0..7 = K kb..kb+7, elems 8..15 = K 16+kb..+7.
__device__ __forceinline__ v16bf load_frag(const unsigned short* __restrict__ p, int ld) {
    int lane = threadIdx.x & 31;
    int m  = lane & 15;
    int kb = (lane >> 4) << 3;            // 0 or 8
    const unsigned short* q = p + (size_t)m * ld + kb;
    Frag32 f;
    f.lo = *reinterpret_cast<const uint4*>(q);
    f.hi = *reinterpret_cast<const uint4*>(q + 16);
    return __builtin_bit_cast(v16bf, f);
}

// ---------------- prep kernels ----------------

__global__ __launch_bounds__(256) void conv_f32_bf16(const float* __restrict__ in,
                                                     unsigned short* __restrict__ out,
                                                     int n4) {
    int i = blockIdx.x * 256 + threadIdx.x;
    if (i >= n4) return;
    float4 v = reinterpret_cast<const float4*>(in)[i];
    ushort4 o;
    o.x = f2bf(v.x); o.y = f2bf(v.y); o.z = f2bf(v.z); o.w = f2bf(v.w);
    reinterpret_cast<ushort4*>(out)[i] = o;
}

// WT[n][k] = bf16(W[k][n]); 32x32 LDS tile transpose
__global__ __launch_bounds__(256) void transpose_w(const float* __restrict__ W,
                                                   unsigned short* __restrict__ WT) {
    __shared__ float tile[32][33];
    int tx = threadIdx.x & 31;
    int ty = threadIdx.x >> 5;            // 0..7
    int bx = blockIdx.x * 32;             // n tile
    int by = blockIdx.y * 32;             // k tile
#pragma unroll
    for (int i = 0; i < 4; ++i)
        tile[ty + 8 * i][tx] = W[(size_t)(by + ty + 8 * i) * HIDDEN + bx + tx];
    __syncthreads();
#pragma unroll
    for (int i = 0; i < 4; ++i)
        WT[(size_t)(bx + ty + 8 * i) * HIDDEN + by + tx] = f2bf(tile[tx][ty + 8 * i]);
}

// ---------------- GEMM with async-LDS double-buffered B tiles ----------------
// Block tile 128x64, 8 waves (each 16x64). B tile (64 cols x 32 k, bf16, 4 KB)
// staged via GLOBAL_LOAD_ASYNC_TO_LDS_B128, 2-stage pipeline on ASYNCcnt.
// mode 0: bf16 Q[b,h,s,d] scaled  mode 1: bf16 K[b,h,s,d]
// mode 2: bf16 V^T[b,h,d,s]       mode 3: fp32 row-major + bias

#define NSTAGE (HIDDEN / 32)   // 32 k-stages

__device__ __forceinline__ void issue_b_tile(unsigned short* lds_dst,          // [64][32]
                                             const unsigned short* __restrict__ Bt,
                                             int col0, int kk) {
    int t = threadIdx.x;
    int n = t >> 2;                        // 0..63  (column within tile)
    int q = t & 3;                         // 16-byte quarter of the 64-byte row
    const unsigned short* g = Bt + (size_t)(col0 + n) * HIDDEN + kk + q * 8;
    unsigned lds_addr = (unsigned)(uintptr_t)(lds_dst + n * 32 + q * 8);
    asm volatile("global_load_async_to_lds_b128 %0, %1, off"
                 :: "v"(lds_addr), "v"((unsigned long long)(uintptr_t)g)
                 : "memory");
}

__global__ __launch_bounds__(256) void gemm_bf16(const unsigned short* __restrict__ A,
                                                 const unsigned short* __restrict__ Bt,
                                                 const float* __restrict__ bias,
                                                 void* __restrict__ outp,
                                                 int mode, float scale) {
    __shared__ unsigned short btile[2][64][32];   // 2 x 4 KB
    int wave = threadIdx.x >> 5;
    int lane = threadIdx.x & 31;
    int row0 = blockIdx.y * 128 + wave * 16;
    int col0 = blockIdx.x * 64;

    v8f acc0{}, acc1{}, acc2{}, acc3{};
    const unsigned short* Arow = A + (size_t)row0 * HIDDEN;

    int m16 = lane & 15;
    int kb  = (lane >> 4) << 3;

    issue_b_tile(&btile[0][0][0], Bt, col0, 0);
    issue_b_tile(&btile[1][0][0], Bt, col0, 32);

    for (int i = 0; i < NSTAGE; ++i) {
        int kk = i * 32;
        if (i < NSTAGE - 1) asm volatile("s_wait_asynccnt 0x1" ::: "memory");
        else                asm volatile("s_wait_asynccnt 0x0" ::: "memory");
        __syncthreads();

        __builtin_prefetch(Arow + kk + 128, 0, 1);
        v16bf a = load_frag(Arow + kk, HIDDEN);

        const unsigned short (*bt)[32] = btile[i & 1];
        v16bf bf[4];
#pragma unroll
        for (int t = 0; t < 4; ++t) {
            const unsigned short* p = &bt[16 * t + m16][kb];
            Frag32 f;
            f.lo = *reinterpret_cast<const uint4*>(p);
            f.hi = *reinterpret_cast<const uint4*>(p + 16);
            bf[t] = __builtin_bit_cast(v16bf, f);
        }
        acc0 = wmma_bf16(a, bf[0], acc0);
        acc1 = wmma_bf16(a, bf[1], acc1);
        acc2 = wmma_bf16(a, bf[2], acc2);
        acc3 = wmma_bf16(a, bf[3], acc3);

        __syncthreads();
        if (i + 2 < NSTAGE)
            issue_b_tile(&btile[i & 1][0][0], Bt, col0, kk + 64);
    }

    int rbase = (lane < 16) ? 0 : 8;      // C/D layout: M = rbase + vgpr
    v8f accs[4] = {acc0, acc1, acc2, acc3};
#pragma unroll
    for (int t = 0; t < 4; ++t) {
        int n = col0 + 16 * t + m16;
        float bv = bias[n];
#pragma unroll
        for (int j = 0; j < 8; ++j) {
            int m = row0 + rbase + j;
            float val = (accs[t][j] + bv) * scale;
            if (mode == 3) {
                reinterpret_cast<float*>(outp)[(size_t)m * HIDDEN + n] = val;
            } else {
                int b = m >> 11, s = m & (SEQ - 1);
                int h = n >> 6,  d = n & (HDIM - 1);
                size_t idx;
                if (mode == 2) idx = ((size_t)(b * HEADS + h) * HDIM + d) * SEQ + s;
                else           idx = ((size_t)(b * HEADS + h) * SEQ + s) * HDIM + d;
                reinterpret_cast<unsigned short*>(outp)[idx] = f2bf(val);
            }
        }
    }
}

// ---------------- flash attention: one wave per 16-query tile ----------------
__global__ __launch_bounds__(256) void attn_kernel(const unsigned short* __restrict__ Q,
                                                   const unsigned short* __restrict__ Km,
                                                   const unsigned short* __restrict__ Vt,
                                                   unsigned short* __restrict__ O2) {
    __shared__ unsigned short plds[8][16 * 40];   // per-wave 16x32 P tile, stride 40
    int wv   = threadIdx.x >> 5;
    int lane = threadIdx.x & 31;
    int wid  = blockIdx.x * 8 + wv;               // 0..4095
    int bh   = wid >> 7;                          // (b*16+h)
    int q0   = (wid & 127) * 16;
    int b    = bh >> 4, h = bh & 15;

    const unsigned short* Qb = Q  + ((size_t)bh * SEQ + q0) * HDIM;
    const unsigned short* Kb = Km + (size_t)bh * SEQ * HDIM;
    const unsigned short* Vb = Vt + (size_t)bh * HDIM * SEQ;

    v16bf aq0 = load_frag(Qb,      HDIM);         // d = 0..31  (scale pre-folded)
    v16bf aq1 = load_frag(Qb + 32, HDIM);         // d = 32..63

    v8f o0{}, o1{}, o2{}, o3{};
    float mrow[8], lrow[8];
#pragma unroll
    for (int j = 0; j < 8; ++j) { mrow[j] = -1e30f; lrow[j] = 0.0f; }
    int rbase = (lane < 16) ? 0 : 8;
    int c0    = lane & 15;
    unsigned short* slab = &plds[wv][0];

    for (int kt = 0; kt < SEQ; kt += 32) {
        __builtin_prefetch(Kb + (size_t)(kt + 32) * HDIM, 0, 1);
        // scores for 32 keys: two 16x16 tiles, K-dim = d (64 -> 2 WMMA each)
        v8f s0{}, s1{};
        s0 = wmma_bf16(aq0, load_frag(Kb + (size_t)kt * HDIM,             HDIM), s0);
        s0 = wmma_bf16(aq1, load_frag(Kb + (size_t)kt * HDIM + 32,        HDIM), s0);
        s1 = wmma_bf16(aq0, load_frag(Kb + (size_t)(kt + 16) * HDIM,      HDIM), s1);
        s1 = wmma_bf16(aq1, load_frag(Kb + (size_t)(kt + 16) * HDIM + 32, HDIM), s1);

        float nm[8], corr[8], p0[8], p1[8];
#pragma unroll
        for (int j = 0; j < 8; ++j) {
            float v = fmaxf(s0[j], s1[j]);        // row max across 16 key lanes
            v = fmaxf(v, __shfl_xor(v, 1, 32));
            v = fmaxf(v, __shfl_xor(v, 2, 32));
            v = fmaxf(v, __shfl_xor(v, 4, 32));
            v = fmaxf(v, __shfl_xor(v, 8, 32));
            nm[j]   = fmaxf(mrow[j], v);
            corr[j] = __expf(mrow[j] - nm[j]);
            p0[j]   = __expf(s0[j] - nm[j]);
            p1[j]   = __expf(s1[j] - nm[j]);
            float rs = p0[j] + p1[j];
            rs += __shfl_xor(rs, 1, 32);
            rs += __shfl_xor(rs, 2, 32);
            rs += __shfl_xor(rs, 4, 32);
            rs += __shfl_xor(rs, 8, 32);
            lrow[j] = lrow[j] * corr[j] + rs;
            mrow[j] = nm[j];
        }
#pragma unroll
        for (int j = 0; j < 8; ++j) {
            o0[j] *= corr[j]; o1[j] *= corr[j]; o2[j] *= corr[j]; o3[j] *= corr[j];
        }
        // C-layout -> A-layout transpose of the P tile through LDS
#pragma unroll
        for (int j = 0; j < 8; ++j) {
            int row = rbase + j;
            slab[row * 40 + c0]      = f2bf(p0[j]);
            slab[row * 40 + c0 + 16] = f2bf(p1[j]);
        }
        asm volatile("s_wait_dscnt 0" ::: "memory");
        int kb = (lane >> 4) << 3;
        const unsigned short* pp = slab + c0 * 40 + kb;
        Frag32 f;
        f.lo = *reinterpret_cast<const uint4*>(pp);
        f.hi = *reinterpret_cast<const uint4*>(pp + 16);
        v16bf ap = __builtin_bit_cast(v16bf, f);
        asm volatile("s_wait_dscnt 0" ::: "memory");

        // O += P(16x32) @ V(32x64); V stored as Vt[d][s] so B frags are contiguous
        o0 = wmma_bf16(ap, load_frag(Vb + (size_t) 0 * SEQ + kt, SEQ), o0);
        o1 = wmma_bf16(ap, load_frag(Vb + (size_t)16 * SEQ + kt, SEQ), o1);
        o2 = wmma_bf16(ap, load_frag(Vb + (size_t)32 * SEQ + kt, SEQ), o2);
        o3 = wmma_bf16(ap, load_frag(Vb + (size_t)48 * SEQ + kt, SEQ), o3);
    }

    // normalize and store in the reference's transpose(0,1,3,2).reshape layout:
    // M2[b][(h*64+d)*2 + q/1024][q%1024] = O[b,h,q,d]
    float inv[8];
#pragma unroll
    for (int j = 0; j < 8; ++j) inv[j] = 1.0f / lrow[j];
    int sHi   = q0 >> 10;
    int cBase = (q0 & 1023) + rbase;
    v8f oo[4] = {o0, o1, o2, o3};
#pragma unroll
    for (int t = 0; t < 4; ++t) {
        int d = 16 * t + c0;
        int r = (h * HDIM + d) * 2 + sHi;
        alignas(16) unsigned short pk[8];
#pragma unroll
        for (int j = 0; j < 8; ++j) pk[j] = f2bf(oo[t][j] * inv[j]);
        *reinterpret_cast<uint4*>(&O2[((size_t)b * SEQ + r) * HIDDEN + cBase]) =
            *reinterpret_cast<const uint4*>(pk);
    }
}

// ---------------- host launcher ----------------
extern "C" void kernel_launch(void* const* d_in, const int* in_sizes, int n_in,
                              void* d_out, int out_size, void* d_ws, size_t ws_size,
                              hipStream_t stream) {
    const float* X  = (const float*)d_in[0];
    const float* wq = (const float*)d_in[1];
    const float* bq = (const float*)d_in[2];
    const float* wk = (const float*)d_in[3];
    const float* bk = (const float*)d_in[4];
    const float* wv = (const float*)d_in[5];
    const float* bv = (const float*)d_in[6];
    const float* wo = (const float*)d_in[7];
    const float* bo = (const float*)d_in[8];

    char* ws = (char*)d_ws;
    const size_t MB = 1024 * 1024;
    unsigned short* Xbf = (unsigned short*)(ws);             // 8 MB
    unsigned short* WqT = (unsigned short*)(ws +  8 * MB);   // 2 MB each
    unsigned short* WkT = (unsigned short*)(ws + 10 * MB);
    unsigned short* WvT = (unsigned short*)(ws + 12 * MB);
    unsigned short* WoT = (unsigned short*)(ws + 14 * MB);
    unsigned short* Qm  = (unsigned short*)(ws + 16 * MB);   // 8 MB each
    unsigned short* Km  = (unsigned short*)(ws + 24 * MB);
    unsigned short* Vt  = (unsigned short*)(ws + 32 * MB);
    unsigned short* O2  = (unsigned short*)(ws + 40 * MB);   // total 48 MB

    conv_f32_bf16<<<(ROWS * HIDDEN / 4) / 256, 256, 0, stream>>>(X, Xbf, ROWS * HIDDEN / 4);

    dim3 tg(HIDDEN / 32, HIDDEN / 32);
    transpose_w<<<tg, 256, 0, stream>>>(wq, WqT);
    transpose_w<<<tg, 256, 0, stream>>>(wk, WkT);
    transpose_w<<<tg, 256, 0, stream>>>(wv, WvT);
    transpose_w<<<tg, 256, 0, stream>>>(wo, WoT);

    dim3 gg(HIDDEN / 64, ROWS / 128);
    const float scale = 1.0f / 32.0f;   // 1/sqrt(HIDDEN), folded into Q (incl. bias)
    gemm_bf16<<<gg, 256, 0, stream>>>(Xbf, WqT, bq, Qm, 0, scale);
    gemm_bf16<<<gg, 256, 0, stream>>>(Xbf, WkT, bk, Km, 1, 1.0f);
    gemm_bf16<<<gg, 256, 0, stream>>>(Xbf, WvT, bv, Vt, 2, 1.0f);

    attn_kernel<<<(BATCH * HEADS * (SEQ / 16)) / 8, 256, 0, stream>>>(Qm, Km, Vt, O2);

    gemm_bf16<<<gg, 256, 0, stream>>>(O2, WoT, bo, d_out, 3, 1.0f);
}
